// LIIF_83915071030150
// MI455X (gfx1250) — matrix operand order repaired and checked by reading, not compiled, not run on hardware
//
#include <hip/hip_runtime.h>
#include <hip/hip_bf16.h>
#include <math.h>

typedef _Float16 half_t;
typedef _Float16 v16h __attribute__((ext_vector_type(16)));
typedef _Float16 v8h  __attribute__((ext_vector_type(8)));
typedef float    v8f  __attribute__((ext_vector_type(8)));

#define BB 2
#define HH 96
#define WW 96
#define QQ 30000
#define CC 64
#define NPIX (BB*HH*WW)          /* 18432 */
#define K0REAL 580
#define K0PAD  608               /* 19 * 32 */
#define K0TILES 19
#define NHID 256
#define NTILES_HID 16
#define KTILES_HID 8
#define BQ (BB*QQ)               /* 60000  */
#define NROWS (4*BQ)             /* 240000 */

/* per-wave row tile: 2 sub-tiles of 16 rows -> 32 rows/wave, 128 rows/block */
#define ROWS_PER_WAVE 32
#define ROWS_PER_BLOCK 128

/* packed fp16 weight layout, offsets in halfs.
 * One fragment = 512 halfs = [lane 0..31][j 0..15]:
 *   n = nt*16 + (lane & 15)
 *   k = kt*32 + ((lane < 16) ? 0 : 16) + j
 * fragments ordered nt-major, kt-minor (so the kt loop streams contiguously). */
#define WP_L0 0
#define WP_L0_SZ (K0TILES*NTILES_HID*512)   /* 155648 */
#define WP_L1 (WP_L0 + WP_L0_SZ)
#define WP_L_SZ (KTILES_HID*NTILES_HID*512) /* 65536  */
#define WP_L2 (WP_L1 + WP_L_SZ)
#define WP_L3 (WP_L2 + WP_L_SZ)
#define WP_L4 (WP_L3 + WP_L_SZ)
#define WP_L4_SZ (KTILES_HID*1*512)         /* 4096   */
#define WP_TOTAL (WP_L4 + WP_L4_SZ)         /* 356352 */

/* ------------------------------------------------------------------ */
/* Weight packing: fp32 (K,N) row-major -> fp16 WMMA-B fragments      */
/* ------------------------------------------------------------------ */
__global__ void pack_w(const float* __restrict__ W, half_t* __restrict__ dst,
                       int Kreal, int Nreal, int Ktiles, int total)
{
    int idx = blockIdx.x * blockDim.x + threadIdx.x;
    if (idx >= total) return;
    int frag   = idx >> 9;
    int within = idx & 511;
    int lane   = within >> 4;
    int j      = within & 15;
    int nt = frag / Ktiles;
    int kt = frag - nt * Ktiles;
    int n = nt * 16 + (lane & 15);
    int k = kt * 32 + ((lane < 16) ? 0 : 16) + j;
    float v = 0.0f;
    if (k < Kreal && n < Nreal) v = W[(size_t)k * Nreal + n];
    dst[idx] = (half_t)v;
}

/* ------------------------------------------------------------------ */
/* Conv 3x3 (3->64, SAME) + bias + tanh, NCHW in -> NHWC fp16 out     */
/* ------------------------------------------------------------------ */
__global__ void __launch_bounds__(256) conv_tanh(const float* __restrict__ inp,
                                                 const float* __restrict__ cw,
                                                 const float* __restrict__ cb,
                                                 half_t* __restrict__ feath)
{
    __shared__ float wsm[CC * 27];
    __shared__ float bsm[CC];
    for (int i = threadIdx.x; i < CC * 27; i += blockDim.x) wsm[i] = cw[i];
    for (int i = threadIdx.x; i < CC;      i += blockDim.x) bsm[i] = cb[i];
    __syncthreads();

    int pix = blockIdx.x * blockDim.x + threadIdx.x;
    if (pix >= NPIX) return;
    int x = pix % WW;
    int y = (pix / WW) % HH;
    int b = pix / (WW * HH);

    float v[27];
#pragma unroll
    for (int ci = 0; ci < 3; ++ci)
#pragma unroll
        for (int ky = 0; ky < 3; ++ky)
#pragma unroll
            for (int kx = 0; kx < 3; ++kx) {
                int yy = y + ky - 1, xx = x + kx - 1;
                float val = 0.0f;
                if (yy >= 0 && yy < HH && xx >= 0 && xx < WW)
                    val = inp[(((size_t)b * 3 + ci) * HH + yy) * WW + xx];
                v[ci * 9 + ky * 3 + kx] = val;
            }

    half_t* out = feath + (size_t)pix * CC;
    for (int c = 0; c < CC; ++c) {
        float acc = bsm[c];
#pragma unroll
        for (int k = 0; k < 27; ++k) acc = fmaf(v[k], wsm[c * 27 + k], acc);
        out[c] = (half_t)tanhf(acc);
    }
}

/* ------------------------------------------------------------------ */
/* Unfold: feath NHWC -> fuh[pix][c*9+p], zero-padded borders         */
/* ------------------------------------------------------------------ */
__global__ void __launch_bounds__(256) im2col(const half_t* __restrict__ feath,
                                              half_t* __restrict__ fuh)
{
    int pix = blockIdx.x * blockDim.x + threadIdx.x;
    if (pix >= NPIX) return;
    int x = pix % WW;
    int y = (pix / WW) % HH;
    int b = pix / (WW * HH);
    half_t* dst = fuh + (size_t)pix * 576;
#pragma unroll
    for (int p = 0; p < 9; ++p) {
        int dy = p / 3 - 1, dx = p % 3 - 1;
        int yy = y + dy, xx = x + dx;
        bool ok = (yy >= 0 && yy < HH && xx >= 0 && xx < WW);
        const half_t* src = feath + ((((size_t)b * HH + (ok ? yy : 0)) * WW + (ok ? xx : 0)) * CC);
        for (int c = 0; c < CC; ++c)
            dst[c * 9 + p] = ok ? src[c] : (half_t)0.0f;
    }
}

/* ------------------------------------------------------------------ */
/* WMMA MLP                                                            */
/* ------------------------------------------------------------------ */
__device__ __forceinline__ v16h load_afrag(const half_t* X, int xstride,
                                           int ln, int hi, int kt)
{
    /* A 16x32 in the contiguous k order used here (same as packed B): lanes 0-15 (row=lane): K {0..15};
       lanes 16-31 (row=lane-16): K {16..31} */
    const v8h* p0 = (const v8h*)(X + (size_t)ln * xstride + kt * 32 + hi * 16);
    const v8h* p1 = (const v8h*)(X + (size_t)ln * xstride + kt * 32 + hi * 16 + 8);
    v8h lo = *p0, hi8 = *p1;
    v16h a;
#pragma unroll
    for (int i = 0; i < 8; ++i) { a[i] = lo[i]; a[8 + i] = hi8[i]; }
    return a;
}

/* one layer over TWO 16-row sub-tiles: each B fragment feeds 2 WMMAs   */
__device__ __forceinline__ void run_layer2(const half_t* __restrict__ X, int xstride,
                                           half_t* __restrict__ Y, int ystride,
                                           const half_t* __restrict__ wfrag,
                                           const float* __restrict__ bias,
                                           int Ktiles, int lane)
{
    const int hi = lane >> 4, ln = lane & 15;
    const half_t* X1 = X + (size_t)16 * xstride;
    for (int nt = 0; nt < NTILES_HID; ++nt) {
        v8f acc0 = {0.f, 0.f, 0.f, 0.f, 0.f, 0.f, 0.f, 0.f};
        v8f acc1 = {0.f, 0.f, 0.f, 0.f, 0.f, 0.f, 0.f, 0.f};
        const half_t* wp = wfrag + ((size_t)nt * Ktiles) * 512 + lane * 16;
        if (nt + 1 < NTILES_HID)
            __builtin_prefetch(wfrag + ((size_t)(nt + 1) * Ktiles) * 512 + lane * 16, 0, 1);
        for (int kt = 0; kt < Ktiles; ++kt) {
            v16h bf = *(const v16h*)wp;
            wp += 512;
            v16h a0 = load_afrag(X,  xstride, ln, hi, kt);
            v16h a1 = load_afrag(X1, xstride, ln, hi, kt);
            acc0 = __builtin_amdgcn_wmma_f32_16x16x32_f16(
                false, a0, false, bf, (short)0, acc0, false, false);
            acc1 = __builtin_amdgcn_wmma_f32_16x16x32_f16(
                false, a1, false, bf, (short)0, acc1, false, false);
        }
        int n = nt * 16 + ln;
        float bv = bias[n];
#pragma unroll
        for (int rr = 0; rr < 8; ++rr) {
            float v0 = fmaxf(acc0[rr] + bv, 0.0f);
            float v1 = fmaxf(acc1[rr] + bv, 0.0f);
            Y[(size_t)(hi * 8 + rr) * ystride + n]        = (half_t)v0;
            Y[(size_t)(16 + hi * 8 + rr) * ystride + n]   = (half_t)v1;
        }
    }
}

__global__ void __launch_bounds__(128) mlp_kernel(
    const float* __restrict__ coord, const float* __restrict__ cell,
    const half_t* __restrict__ fuh,  const half_t* __restrict__ wpack,
    const float* __restrict__ b0, const float* __restrict__ b1,
    const float* __restrict__ b2, const float* __restrict__ b3,
    const float* __restrict__ b4,
    float* __restrict__ yout, float* __restrict__ areas)
{
    extern __shared__ half_t smem[];
    const int lane = threadIdx.x & 31;
    const int wave = threadIdx.x >> 5;
    half_t* Xa = smem + (size_t)wave * (ROWS_PER_WAVE * K0PAD + ROWS_PER_WAVE * NHID);
    half_t* Xb = Xa + ROWS_PER_WAVE * K0PAD;
    const int rowBase = blockIdx.x * ROWS_PER_BLOCK + wave * ROWS_PER_WAVE;

    /* ---- gather phase: two 16-row sub-tiles; lanes 0..15 own one row ---- */
    for (int st = 0; st < 2; ++st) {
        half_t* Xs = Xa + (size_t)st * 16 * K0PAD;
        int srcOff = 0;
        if (lane < 16) {
            int r = rowBase + st * 16 + lane;       /* r = s*B*Q + b*Q + q */
            int s = r / BQ;
            int t = r - s * BQ;                     /* b*Q + q */
            int bb = t / QQ;
            float c0  = coord[(size_t)t * 2 + 0];
            float c1  = coord[(size_t)t * 2 + 1];
            float ce0 = cell [(size_t)t * 2 + 0];
            float ce1 = cell [(size_t)t * 2 + 1];
            float oy = (s >> 1) ? 1.0f : -1.0f;
            float ox = (s & 1)  ? 1.0f : -1.0f;
            float e0 = c0 + oy * (1.0f / HH) + 1e-6f;
            float e1 = c1 + ox * (1.0f / WW) + 1e-6f;
            const float lim = 1.0f - 1e-10f;
            e0 = fminf(fmaxf(e0, -lim), lim);
            e1 = fminf(fmaxf(e1, -lim), lim);
            int iy = (int)rintf((e0 + 1.0f) * (0.5f * HH) - 0.5f);
            int ix = (int)rintf((e1 + 1.0f) * (0.5f * WW) - 0.5f);
            iy = iy < 0 ? 0 : (iy > HH - 1 ? HH - 1 : iy);
            ix = ix < 0 ? 0 : (ix > WW - 1 ? WW - 1 : ix);
            float qc0 = -1.0f + (2.0f * iy + 1.0f) / HH;
            float qc1 = -1.0f + (2.0f * ix + 1.0f) / WW;
            float rel0 = (c0 - qc0) * HH;
            float rel1 = (c1 - qc1) * WW;
            areas[r] = fabsf(rel0 * rel1) + 1e-9f;
            half_t* xr = Xs + (size_t)lane * K0PAD;
            xr[576] = (half_t)rel0;
            xr[577] = (half_t)rel1;
            xr[578] = (half_t)(ce0 * HH);
            xr[579] = (half_t)(ce1 * WW);
#pragma unroll
            for (int k = K0REAL; k < K0PAD; ++k) xr[k] = (half_t)0.0f;
            srcOff = ((bb * HH + iy) * WW + ix) * 576;
        }
        /* cooperative copy: 16 rows x 1152 B (72 uint4) from fuh into LDS */
        for (int i = 0; i < 16; ++i) {
            int so = __shfl(srcOff, i, 32);
            const uint4* src = (const uint4*)(fuh + (size_t)so);
            uint4* dst = (uint4*)(Xs + (size_t)i * K0PAD);
            for (int j = lane; j < 72; j += 32) dst[j] = src[j];
        }
    }
    __syncthreads();

    run_layer2(Xa, K0PAD, Xb, NHID, wpack + WP_L0, b0, K0TILES,    lane);
    __syncthreads();
    run_layer2(Xb, NHID, Xa, K0PAD, wpack + WP_L1, b1, KTILES_HID, lane);
    __syncthreads();
    run_layer2(Xa, K0PAD, Xb, NHID, wpack + WP_L2, b2, KTILES_HID, lane);
    __syncthreads();
    run_layer2(Xb, NHID, Xa, K0PAD, wpack + WP_L3, b3, KTILES_HID, lane);
    __syncthreads();

    /* ---- layer 4: 256 -> 3 (N padded to 16), both sub-tiles ---- */
    {
        const int hi = lane >> 4, ln = lane & 15;
        v8f acc0 = {0.f, 0.f, 0.f, 0.f, 0.f, 0.f, 0.f, 0.f};
        v8f acc1 = {0.f, 0.f, 0.f, 0.f, 0.f, 0.f, 0.f, 0.f};
        const half_t* wp = wpack + WP_L4 + lane * 16;
        const half_t* X1 = Xa + (size_t)16 * K0PAD;
        for (int kt = 0; kt < KTILES_HID; ++kt) {
            v16h bf = *(const v16h*)wp;
            wp += 512;
            v16h a0 = load_afrag(Xa, K0PAD, ln, hi, kt);
            v16h a1 = load_afrag(X1, K0PAD, ln, hi, kt);
            acc0 = __builtin_amdgcn_wmma_f32_16x16x32_f16(
                false, a0, false, bf, (short)0, acc0, false, false);
            acc1 = __builtin_amdgcn_wmma_f32_16x16x32_f16(
                false, a1, false, bf, (short)0, acc1, false, false);
        }
        if (ln < 3) {
            float bv = b4[ln];
#pragma unroll
            for (int rr = 0; rr < 8; ++rr) {
                int row0 = rowBase + hi * 8 + rr;
                int row1 = rowBase + 16 + hi * 8 + rr;
                yout[(size_t)row0 * 3 + ln] = acc0[rr] + bv;
                yout[(size_t)row1 * 3 + ln] = acc1[rr] + bv;
            }
        }
    }
}

/* ------------------------------------------------------------------ */
/* Area-weighted combine over the 4 offsets + tanh * 1.01             */
/* ------------------------------------------------------------------ */
__global__ void __launch_bounds__(256) combine_kernel(const float* __restrict__ yin,
                                                      const float* __restrict__ areas,
                                                      float* __restrict__ out)
{
    int i = blockIdx.x * blockDim.x + threadIdx.x;   /* i = b*Q + q */
    if (i >= BQ) return;
    float a0 = areas[i];
    float a1 = areas[BQ + i];
    float a2 = areas[2 * BQ + i];
    float a3 = areas[3 * BQ + i];
    float inv = 1.0f / (a0 + a1 + a2 + a3);
    float w0 = a3 * inv, w1 = a2 * inv, w2 = a1 * inv, w3 = a0 * inv; /* areas[::-1] */
#pragma unroll
    for (int j = 0; j < 3; ++j) {
        float acc = yin[(size_t)i * 3 + j] * w0
                  + yin[((size_t)BQ + i) * 3 + j] * w1
                  + yin[((size_t)2 * BQ + i) * 3 + j] * w2
                  + yin[((size_t)3 * BQ + i) * 3 + j] * w3;
        out[(size_t)i * 3 + j] = tanhf(acc) * 1.01f;
    }
}

/* ------------------------------------------------------------------ */
extern "C" void kernel_launch(void* const* d_in, const int* in_sizes, int n_in,
                              void* d_out, int out_size, void* d_ws, size_t ws_size,
                              hipStream_t stream)
{
    (void)in_sizes; (void)n_in; (void)out_size; (void)ws_size;
    const float* inp    = (const float*)d_in[0];
    const float* coord  = (const float*)d_in[1];
    const float* cell   = (const float*)d_in[2];
    const float* conv_w = (const float*)d_in[3];
    const float* conv_b = (const float*)d_in[4];
    const float* w0 = (const float*)d_in[5];
    const float* b0 = (const float*)d_in[6];
    const float* w1 = (const float*)d_in[7];
    const float* b1 = (const float*)d_in[8];
    const float* w2 = (const float*)d_in[9];
    const float* b2 = (const float*)d_in[10];
    const float* w3 = (const float*)d_in[11];
    const float* b3 = (const float*)d_in[12];
    const float* w4 = (const float*)d_in[13];
    const float* b4 = (const float*)d_in[14];

    /* workspace layout (all sizes multiples of 256 B) */
    char* ws = (char*)d_ws;
    size_t off = 0;
    half_t* feath = (half_t*)(ws + off); off += (size_t)NPIX * CC * sizeof(half_t);     /* 2,359,296 */
    half_t* fuh   = (half_t*)(ws + off); off += (size_t)NPIX * 576 * sizeof(half_t);    /* 21,233,664 */
    half_t* wpack = (half_t*)(ws + off); off += (size_t)WP_TOTAL * sizeof(half_t);      /* 712,704 */
    float*  yout  = (float* )(ws + off); off += (size_t)NROWS * 3 * sizeof(float);      /* 2,880,000 */
    float*  areas = (float* )(ws + off); off += (size_t)NROWS * sizeof(float);          /* 960,000 */

    /* 1) pack weights to fp16 WMMA-B fragment layout */
    pack_w<<<(WP_L0_SZ + 255) / 256, 256, 0, stream>>>(w0, wpack + WP_L0, K0REAL, NHID, K0TILES, WP_L0_SZ);
    pack_w<<<(WP_L_SZ  + 255) / 256, 256, 0, stream>>>(w1, wpack + WP_L1, NHID,   NHID, KTILES_HID, WP_L_SZ);
    pack_w<<<(WP_L_SZ  + 255) / 256, 256, 0, stream>>>(w2, wpack + WP_L2, NHID,   NHID, KTILES_HID, WP_L_SZ);
    pack_w<<<(WP_L_SZ  + 255) / 256, 256, 0, stream>>>(w3, wpack + WP_L3, NHID,   NHID, KTILES_HID, WP_L_SZ);
    pack_w<<<(WP_L4_SZ + 255) / 256, 256, 0, stream>>>(w4, wpack + WP_L4, NHID,   3,    KTILES_HID, WP_L4_SZ);

    /* 2) conv + tanh -> fp16 NHWC feature map */
    conv_tanh<<<NPIX / 256, 256, 0, stream>>>(inp, conv_w, conv_b, feath);

    /* 3) unfold into contiguous 576-half records */
    im2col<<<NPIX / 256, 256, 0, stream>>>(feath, fuh);

    /* 4) WMMA MLP: 128 rows/block, 32 rows per wave32 (2 sub-tiles, B reused 2x) */
    size_t lds_bytes = (size_t)4 * (ROWS_PER_WAVE * K0PAD + ROWS_PER_WAVE * NHID)
                       * sizeof(half_t);                                   /* 221,184 */
    mlp_kernel<<<NROWS / ROWS_PER_BLOCK, 128, lds_bytes, stream>>>(
        coord, cell, fuh, wpack, b0, b1, b2, b3, b4, yout, areas);

    /* 5) combine 4 offsets -> output (B, Q, 3) */
    combine_kernel<<<(BQ + 255) / 256, 256, 0, stream>>>(yout, areas, (float*)d_out);
}
